// MoralFrameworkGraphLayer_63101659513102
// MI455X (gfx1250) — compile-verified, matrix operand order checked by reading, not run on hardware
//
#include <hip/hip_runtime.h>
#include <math.h>

typedef __attribute__((ext_vector_type(16))) __bf16 v16bf;
typedef __attribute__((ext_vector_type(8)))  float  v8f;

#define IN_C  128
#define OUT_C 128

// ---------------- degree / normalization ----------------

__global__ void k_deg_init(float* __restrict__ deg, int n) {
    int i = blockIdx.x * blockDim.x + threadIdx.x;
    if (i < n) deg[i] = 1.0f;  // self-loop contributes 1 to every node's degree
}

__global__ void k_deg_edges(const int* __restrict__ col, float* __restrict__ deg, int e) {
    int i = blockIdx.x * blockDim.x + threadIdx.x;
    if (i < e) atomicAdd(&deg[col[i]], 1.0f);
}

__global__ void k_dinv(float* __restrict__ deg, int n) {
    int i = blockIdx.x * blockDim.x + threadIdx.x;
    if (i < n) deg[i] = rsqrtf(deg[i]);  // in place: deg -> deg^{-1/2}
}

// ---------------- h = x @ W^T via bf16 WMMA (3-product fp32 recovery) ----------------
// One wave per 16x16 output tile. Block = 8 waves = one 16-row strip across all
// 8 column tiles (128 outputs). A-fragment: lane l holds row m0+(l&15); element e
// maps to k = 8*(l>>4) + (e<8 ? e : e+8). B symmetric with N in place of M.

__global__ __launch_bounds__(256)
void k_gemm_wmma(const float* __restrict__ x, const float* __restrict__ W,
                 float* __restrict__ h, int n_nodes) {
    const int wave = threadIdx.x >> 5;
    const int lane = threadIdx.x & 31;
    const int lrow = lane & 15;
    const int grp  = lane >> 4;
    const int m0   = blockIdx.x * 16;
    const int n0   = wave * 16;

    int arow = m0 + lrow;
    if (arow > n_nodes - 1) arow = n_nodes - 1;       // clamp loads (uniform-tile safe)
    const float* __restrict__ xr = x + (long)arow * IN_C;
    const float* __restrict__ wr = W + (long)(n0 + lrow) * IN_C;

    v8f acc = {0.f, 0.f, 0.f, 0.f, 0.f, 0.f, 0.f, 0.f};

#pragma unroll
    for (int k0 = 0; k0 < IN_C; k0 += 32) {
        v16bf a_hi, a_lo, b_hi, b_lo;
#pragma unroll
        for (int e = 0; e < 16; ++e) {
            const int rk = k0 + ((e & 8) ? (e + 8) : e) + 8 * grp;
            const float av = xr[rk];
            const __bf16 ah = (__bf16)av;
            a_hi[e] = ah;
            a_lo[e] = (__bf16)(av - (float)ah);
            const float bv = wr[rk];
            const __bf16 bh = (__bf16)bv;
            b_hi[e] = bh;
            b_lo[e] = (__bf16)(bv - (float)bh);
        }
        // fp32-accurate product: hi*hi + lo*hi + hi*lo (lo*lo ~ 2^-16 rel, dropped)
        acc = __builtin_amdgcn_wmma_f32_16x16x32_bf16(false, a_hi, false, b_hi,
                                                      (short)0, acc, false, false);
        acc = __builtin_amdgcn_wmma_f32_16x16x32_bf16(false, a_lo, false, b_hi,
                                                      (short)0, acc, false, false);
        acc = __builtin_amdgcn_wmma_f32_16x16x32_bf16(false, a_hi, false, b_lo,
                                                      (short)0, acc, false, false);
    }

    // C/D layout: VGPR r, lanes 0-15 -> M=r; lanes 16-31 -> M=8+r; N = lane%16
    float* __restrict__ hr = h + n0 + lrow;
#pragma unroll
    for (int r = 0; r < 8; ++r) {
        const int m = m0 + r + 8 * grp;
        if (m < n_nodes) hr[(long)m * OUT_C] = acc[r];
    }
}

// ---------------- self-loop init: out = h * dinv^2 ----------------

__global__ void k_self_init(const float* __restrict__ h, const float* __restrict__ dinv,
                            float* __restrict__ out, long total) {
    long idx = (long)blockIdx.x * blockDim.x + threadIdx.x;
    if (idx >= total) return;
    const int i = (int)(idx >> 7);  // OUT_C == 128
    const float di = dinv[i];
    out[idx] = h[idx] * di * di;
}

// ---------------- edge scatter: one wave per edge, float4 per lane ----------------

__global__ __launch_bounds__(256)
void k_scatter(const int* __restrict__ ei, const float* __restrict__ h,
               const float* __restrict__ dinv, float* __restrict__ out, int n_edges) {
    const long t = (long)blockIdx.x * blockDim.x + threadIdx.x;
    const int e    = (int)(t >> 5);
    const int lane = (int)(t & 31);
    if (e >= n_edges) return;
    const int row = ei[e];            // source
    const int col = ei[n_edges + e];  // target
    const float nrm = dinv[row] * dinv[col];
    const float4 v = reinterpret_cast<const float4*>(h + (long)row * OUT_C)[lane];
    float* dst = out + (long)col * OUT_C + lane * 4;
    atomicAdd(dst + 0, v.x * nrm);
    atomicAdd(dst + 1, v.y * nrm);
    atomicAdd(dst + 2, v.z * nrm);
    atomicAdd(dst + 3, v.w * nrm);
}

// ---------------- finalize: out = silu(out * dinv + b) ----------------

__global__ void k_finalize(float* __restrict__ out, const float* __restrict__ dinv,
                           const float* __restrict__ b, long total) {
    long idx = (long)blockIdx.x * blockDim.x + threadIdx.x;
    if (idx >= total) return;
    const int i = (int)(idx >> 7);
    const int c = (int)(idx & 127);
    const float v = out[idx] * dinv[i] + b[c];
    out[idx] = v / (1.0f + __expf(-v));
}

// ---------------- launcher ----------------

extern "C" void kernel_launch(void* const* d_in, const int* in_sizes, int n_in,
                              void* d_out, int out_size, void* d_ws, size_t ws_size,
                              hipStream_t stream) {
    const float* x  = (const float*)d_in[0];
    const int*   ei = (const int*)d_in[1];
    const float* W  = (const float*)d_in[2];
    const float* b  = (const float*)d_in[3];
    float* out = (float*)d_out;

    const int n_nodes = in_sizes[0] / IN_C;
    const int n_edges = in_sizes[1] / 2;
    const long total  = (long)n_nodes * OUT_C;

    // workspace: dinv[n_nodes] then h[n_nodes*OUT_C] (offset padded to 128 floats)
    float* dinv = (float*)d_ws;
    float* h    = dinv + (((long)n_nodes + 127) & ~127L);
    const int* col = ei + n_edges;

    const int B = 256;
    k_deg_init <<<(n_nodes + B - 1) / B, B, 0, stream>>>(dinv, n_nodes);
    k_deg_edges<<<(n_edges + B - 1) / B, B, 0, stream>>>(col, dinv, n_edges);
    k_dinv     <<<(n_nodes + B - 1) / B, B, 0, stream>>>(dinv, n_nodes);

    k_gemm_wmma<<<(n_nodes + 15) / 16, 256, 0, stream>>>(x, W, h, n_nodes);

    k_self_init<<<(int)((total + B - 1) / B), B, 0, stream>>>(h, dinv, out, total);
    k_scatter  <<<(int)(((long)n_edges * 32 + B - 1) / B), B, 0, stream>>>(ei, h, dinv, out, n_edges);
    k_finalize <<<(int)((total + B - 1) / B), B, 0, stream>>>(out, dinv, b, total);
}
